// BilinearAttention_6339371729520
// MI455X (gfx1250) — matrix operand order repaired and checked by reading, not compiled, not run on hardware
//
#include <hip/hip_runtime.h>
#include <hip/hip_fp16.h>

typedef __attribute__((ext_vector_type(16))) _Float16 v16h;
typedef __attribute__((ext_vector_type(8)))  _Float16 v8h;
typedef __attribute__((ext_vector_type(8)))  float    v8f;

// Problem constants (match reference setup_inputs)
#define BB 8
#define CC 512
#define NN 2304     // 48*48
#define PP 64
#define MM (BB * NN) // 18432

#define LDS_STRIDE 40   // 32 halfs + 8 pad; 80B row stride keeps 16B alignment

__device__ __forceinline__ v8f wmma_f16(v16h a, v16h b, v8f c) {
  // D = A(16x32 f16) * B(32x16 f16) + C(16x16 f32)
  return __builtin_amdgcn_wmma_f32_16x16x32_f16(false, a, false, b, (short)0, c, false, false);
}

// A fragment: 16x32 tile, row-major (ld elements per row).
// Lane l<16: M=l, K in {0..7, 16..23}; lane>=16: M=l-16, K in {8..15, 24..31}.
__device__ __forceinline__ v16h load_a_frag(const _Float16* __restrict__ base, int ld, int lane) {
  const int m  = lane & 15;
  const int kb = (lane >> 4) << 3;  // 0 or 8
  const _Float16* p = base + (size_t)m * ld + kb;
  v8h lo = *(const v8h*)(p);
  v8h hi = *(const v8h*)(p + 16);
  v16h a;
#pragma unroll
  for (int i = 0; i < 8; ++i) { a[i] = lo[i]; a[i + 8] = hi[i]; }
  return a;
}

// B fragment from B^T stored row-major: Bt[col][k], ld elements per row.
// Lane l<16: N=l, K in {0..7, 16..23}; lane>=16: N=l-16, K in {8..15, 24..31}.
__device__ __forceinline__ v16h load_bt_frag(const _Float16* __restrict__ base, int ld, int lane) {
  const int n  = lane & 15;
  const int kb = (lane >> 4) << 3;  // 0 or 8
  const _Float16* p = base + (size_t)n * ld + kb;
  v8h lo = *(const v8h*)(p);
  v8h hi = *(const v8h*)(p + 16);
  v16h b;
#pragma unroll
  for (int i = 0; i < 8; ++i) { b[i] = lo[i]; b[i + 8] = hi[i]; }
  return b;
}

// ---------------------------------------------------------------- async copy
// CDNA5 async global->LDS: per-lane 16B transfer, INST_OFFSET applies to both
// the global and LDS addresses; tracked by ASYNCcnt.
__device__ __forceinline__ void async_stage32(const _Float16* __restrict__ g, void* lds) {
  unsigned off = (unsigned)(unsigned long long)lds;  // low 32 bits = LDS byte offset
  asm volatile("global_load_async_to_lds_b128 %0, %1, off\n\t"
               "global_load_async_to_lds_b128 %0, %1, off offset:16"
               :: "v"(off), "v"(g) : "memory");
}
__device__ __forceinline__ void async_wait0() {
  asm volatile("s_wait_asynccnt 0x0" ::: "memory");
}

// ---------------------------------------------------------------- pack / cvt

__global__ __launch_bounds__(256) void pack_x_kernel(const float* __restrict__ x,
                                                     _Float16* __restrict__ px) {
  __shared__ float tile[32][33];
  const int b  = blockIdx.z;
  const int n0 = blockIdx.x * 32, c0 = blockIdx.y * 32;
  const int tx = threadIdx.x & 31, ty = threadIdx.x >> 5;  // ty 0..7
  const float* xb = x + (size_t)b * CC * NN;
#pragma unroll
  for (int i = 0; i < 4; ++i)
    tile[ty + i * 8][tx] = xb[(size_t)(c0 + ty + i * 8) * NN + n0 + tx];
  __syncthreads();
  _Float16* pb = px + (size_t)b * NN * CC;
#pragma unroll
  for (int i = 0; i < 4; ++i)
    pb[(size_t)(n0 + ty + i * 8) * CC + c0 + tx] = (_Float16)tile[tx][ty + i * 8];
}

__global__ void cvt_f16(const float* __restrict__ in, _Float16* __restrict__ out, int n) {
  int i = blockIdx.x * blockDim.x + threadIdx.x;
  if (i < n) out[i] = (_Float16)in[i];
}

// ---------------------------------------------------------------- q/k kernel
// qs[m][p] = f16( tanh(px @ Wq^T)[m][p] * aw[p] );  k[m][p] = f16( tanh(px @ Wk^T)[m][p] )
__global__ __launch_bounds__(128) void qk_kernel(const _Float16* __restrict__ px,
                                                 const _Float16* __restrict__ wq,
                                                 const _Float16* __restrict__ wk,
                                                 const float* __restrict__ aw,
                                                 _Float16* __restrict__ qs,
                                                 _Float16* __restrict__ kk) {
  const int lane = threadIdx.x & 31;
  const int wave = threadIdx.x >> 5;      // 4 waves -> p tiles 0..3 (P=64)
  const int row0 = blockIdx.x * 16;
  const int p0   = wave * 16;
  v8f aq = {}, ak = {};
  const _Float16* Ap = px + (size_t)row0 * CC;
  const _Float16* Bq = wq + (size_t)p0 * CC;   // Wq row-major [P][C] == (Wq^T)^T
  const _Float16* Bk = wk + (size_t)p0 * CC;
  for (int k0 = 0; k0 < CC; k0 += 32) {
    v16h a  = load_a_frag(Ap + k0, CC, lane);
    v16h bq = load_bt_frag(Bq + k0, CC, lane);
    v16h bk = load_bt_frag(Bk + k0, CC, lane);
    aq = wmma_f16(a, bq, aq);
    ak = wmma_f16(a, bk, ak);
  }
  const int nn = lane & 15, mb = (lane >> 4) << 3;
  const int pcol = p0 + nn;
  const float w = aw[pcol];
#pragma unroll
  for (int r = 0; r < 8; ++r) {
    size_t row = (size_t)(row0 + mb + r);
    qs[row * PP + pcol] = (_Float16)(tanhf(aq[r]) * w);
    kk[row * PP + pcol] = (_Float16)(tanhf(ak[r]));
  }
}

// ---------------------------------------------------------------- main GEMM
enum { EPI_VT = 0, EPI_EXP = 1, EPI_AV = 2, EPI_FC = 3 };

struct GemmP {
  const _Float16* A;  int lda;   // A row-major [M][K]
  const _Float16* Bt; int ldb;   // B^T row-major [cols][K]
  int K;
  _Float16* outH; int ldo;       // f16 row-major output (EXP: attnexp, AV: c16)
  _Float16* outT;                // EPI_VT: vT16 full buffer, idx (b*C+col)*N + n
  float*    outF;                // EPI_FC: d_out, idx (b*C+col)*N + n
  const float* rs;               // EPI_AV: rowsum[N]
  const float* xres;             // EPI_AV: residual x + b*C*N, idx col*N + row
  const float* bias;             // EPI_FC
  float* bnSum; float* bnSqs;    // EPI_FC atomics, length C
  float* rowsum;                 // EPI_EXP atomics, length N
};

// Block tile 128(M) x 128(cols); 8 waves (4 row-groups x 2 col-groups);
// wave tile 32x64 (2 A-frags x 4 B-frags = 8 accumulators).
// K staged through double-buffered LDS in steps of 32 via async global->LDS.
template <int EPI>
__global__ __launch_bounds__(256) void wmma_gemm_k(GemmP p) {
  __shared__ _Float16 sA[2][128 * LDS_STRIDE];
  __shared__ _Float16 sB[2][128 * LDS_STRIDE];
  __shared__ float red0[128];
  __shared__ float red1[128];

  const int tid  = threadIdx.x;
  const int lane = tid & 31, wave = tid >> 5;
  const int wr = wave >> 1, wc = wave & 1;   // 4 row-groups x 2 col-groups
  const int rowBase = blockIdx.y * 128 + wr * 32;
  const int colBase = blockIdx.x * 128 + wc * 64;

  if (EPI == EPI_EXP) { if (tid < 128) red0[tid] = 0.f; }
  if (EPI == EPI_FC)  { if (tid < 128) { red0[tid] = 0.f; red1[tid] = 0.f; } }

  // staging assignment: each thread owns one 16-half (32B) segment
  const int sr = tid >> 1;          // 0..127
  const int sc = (tid & 1) << 4;    // 0 or 16
  const _Float16* Ag = p.A  + (size_t)(blockIdx.y * 128) * p.lda + (size_t)sr * p.lda + sc;
  const _Float16* Bg = p.Bt + (size_t)(blockIdx.x * 128) * p.ldb + (size_t)sr * p.ldb + sc;
  const int sOff = sr * LDS_STRIDE + sc;

  async_stage32(Ag, &sA[0][sOff]);
  async_stage32(Bg, &sB[0][sOff]);
  async_wait0();
  __syncthreads();

  v8f acc[2][4] = {};
  const int nk = p.K >> 5;
  for (int kt = 0; kt < nk; ++kt) {
    const int cur = kt & 1, nxt = cur ^ 1;
    const bool more = (kt + 1) < nk;
    if (more) {
      async_stage32(Ag + ((kt + 1) << 5), &sA[nxt][sOff]);
      async_stage32(Bg + ((kt + 1) << 5), &sB[nxt][sOff]);
      if (kt + 2 < nk) {  // cover latency of the global->LDS stream
        __builtin_prefetch(Ag + ((kt + 2) << 5), 0, 1);
        __builtin_prefetch(Bg + ((kt + 2) << 5), 0, 1);
      }
    }

    const _Float16* cA = &sA[cur][wr * 32 * LDS_STRIDE];
    const _Float16* cB = &sB[cur][wc * 64 * LDS_STRIDE];
    v16h a0 = load_a_frag(cA, LDS_STRIDE, lane);
    v16h a1 = load_a_frag(cA + 16 * LDS_STRIDE, LDS_STRIDE, lane);
#pragma unroll
    for (int tj = 0; tj < 4; ++tj) {
      v16h bf = load_bt_frag(cB + tj * 16 * LDS_STRIDE, LDS_STRIDE, lane);
      acc[0][tj] = wmma_f16(a0, bf, acc[0][tj]);
      acc[1][tj] = wmma_f16(a1, bf, acc[1][tj]);
    }

    async_wait0();      // next-buffer staging complete before the barrier
    __syncthreads();
  }

  const int nn = lane & 15, mb = (lane >> 4) << 3;

#pragma unroll
  for (int ti = 0; ti < 2; ++ti) {
    const int row0 = rowBase + ti * 16 + mb;
    if (EPI == EPI_EXP) {
      // exp(scores), store f16, accumulate row sums (rows local to this block)
#pragma unroll
      for (int r = 0; r < 8; ++r) {
        const int row = row0 + r;
        float esum = 0.f;
#pragma unroll
        for (int tj = 0; tj < 4; ++tj) {
          float e = __expf(acc[ti][tj][r]);
          p.outH[(size_t)row * p.ldo + colBase + tj * 16 + nn] = (_Float16)e;
          esum += e;
        }
        atomicAdd(&red0[wr * 32 + ti * 16 + mb + r], esum);
      }
    } else {
#pragma unroll
      for (int tj = 0; tj < 4; ++tj) {
        const int col = colBase + tj * 16 + nn;
        if (EPI == EPI_VT) {
          const int b = row0 / NN;  // uniform: 128-row blocks never straddle batches
#pragma unroll
          for (int r = 0; r < 8; ++r) {
            const int n = row0 + r - b * NN;
            p.outT[((size_t)(b * CC + col)) * NN + n] = (_Float16)acc[ti][tj][r];
          }
        } else if (EPI == EPI_AV) {
#pragma unroll
          for (int r = 0; r < 8; ++r) {
            const int row = row0 + r;
            const float t = acc[ti][tj][r] * (1.0f / p.rs[row]) +
                            p.xres[(size_t)col * NN + row];
            p.outH[(size_t)row * p.ldo + col] = (_Float16)t;
          }
        } else {  // EPI_FC
          const int b = row0 / NN;
          const float bias = p.bias[col];
          float s = 0.f, sq = 0.f;
#pragma unroll
          for (int r = 0; r < 8; ++r) {
            const int n = row0 + r - b * NN;
            const float t = acc[ti][tj][r] + bias;
            p.outF[((size_t)(b * CC + col)) * NN + n] = t;
            s += t; sq += t * t;
          }
          atomicAdd(&red0[wc * 64 + tj * 16 + nn], s);
          atomicAdd(&red1[wc * 64 + tj * 16 + nn], sq);
        }
      }
    }
  }

  if (EPI == EPI_EXP) {
    __syncthreads();
    if (tid < 128) atomicAdd(&p.rowsum[blockIdx.y * 128 + tid], red0[tid]);
  }
  if (EPI == EPI_FC) {
    __syncthreads();
    if (tid < 128) {
      atomicAdd(&p.bnSum[blockIdx.x * 128 + tid], red0[tid]);
      atomicAdd(&p.bnSqs[blockIdx.x * 128 + tid], red1[tid]);
    }
  }
}

// ---------------------------------------------------------------- batchnorm

__global__ void bn_finalize(const float* __restrict__ sum, const float* __restrict__ sqs,
                            const float* __restrict__ gamma, const float* __restrict__ beta,
                            float* __restrict__ scale, float* __restrict__ shift,
                            float invCount) {
  int c = blockIdx.x * blockDim.x + threadIdx.x;
  if (c < CC) {
    float m = sum[c] * invCount;
    float v = sqs[c] * invCount - m * m;
    float r = rsqrtf(v + 1e-5f);
    float sc = gamma[c] * r;
    scale[c] = sc;
    shift[c] = beta[c] - m * sc;
  }
}

__global__ void bn_apply(float* __restrict__ out, const float* __restrict__ scale,
                         const float* __restrict__ shift, size_t total) {
  size_t i = blockIdx.x * (size_t)blockDim.x + threadIdx.x;
  if (i < total) {
    int c = (int)((i / NN) % CC);
    float v = out[i] * scale[c] + shift[c];
    out[i] = v > 0.f ? v : 0.f;
  }
}

// ---------------------------------------------------------------- launch

extern "C" void kernel_launch(void* const* d_in, const int* in_sizes, int n_in,
                              void* d_out, int out_size, void* d_ws, size_t ws_size,
                              hipStream_t stream) {
  const float* x     = (const float*)d_in[0];
  const float* Wq    = (const float*)d_in[1];
  const float* Wk    = (const float*)d_in[2];
  const float* Wv    = (const float*)d_in[3];
  const float* aw    = (const float*)d_in[4];
  const float* Wfc   = (const float*)d_in[5];
  const float* bfc   = (const float*)d_in[6];
  const float* gamma = (const float*)d_in[7];
  const float* beta  = (const float*)d_in[8];
  float* out = (float*)d_out;

  char* wsp = (char*)d_ws;
  auto alloc = [&](size_t bytes) -> char* {
    char* p = wsp; wsp += (bytes + 255) & ~(size_t)255; return p;
  };
  _Float16* px16    = (_Float16*)alloc((size_t)MM * CC * 2);   // [B*N][C]
  _Float16* wq16    = (_Float16*)alloc((size_t)PP * CC * 2);   // [P][C]
  _Float16* wk16    = (_Float16*)alloc((size_t)PP * CC * 2);
  _Float16* wv16    = (_Float16*)alloc((size_t)CC * CC * 2);   // [C][C]
  _Float16* wfc16   = (_Float16*)alloc((size_t)CC * CC * 2);
  _Float16* qs16    = (_Float16*)alloc((size_t)MM * PP * 2);   // [B*N][P] (tanh * aw)
  _Float16* k16     = (_Float16*)alloc((size_t)MM * PP * 2);   // [B*N][P] (tanh)
  _Float16* vT16    = (_Float16*)alloc((size_t)BB * CC * NN * 2); // [B][C][N]
  _Float16* attnexp = (_Float16*)alloc((size_t)NN * NN * 2);   // per-batch reuse
  _Float16* c16     = (_Float16*)alloc((size_t)MM * CC * 2);   // content (post-residual)
  float*    rowsum  = (float*)alloc((size_t)NN * 4);           // per-batch reuse
  float*    bnSum   = (float*)alloc((size_t)2 * CC * 4);       // sum | sumsq adjacent
  float*    bnSqs   = bnSum + CC;
  float*    bnScale = (float*)alloc((size_t)2 * CC * 4);
  float*    bnShift = bnScale + CC;

  // 1. pack x -> px16 (transpose [B][C][N] -> [B][N][C], f16), weights -> f16
  pack_x_kernel<<<dim3(NN / 32, CC / 32, BB), 256, 0, stream>>>(x, px16);
  cvt_f16<<<(PP * CC + 255) / 256, 256, 0, stream>>>(Wq, wq16, PP * CC);
  cvt_f16<<<(PP * CC + 255) / 256, 256, 0, stream>>>(Wk, wk16, PP * CC);
  cvt_f16<<<(CC * CC + 255) / 256, 256, 0, stream>>>(Wv, wv16, CC * CC);
  cvt_f16<<<(CC * CC + 255) / 256, 256, 0, stream>>>(Wfc, wfc16, CC * CC);

  // 2. q/k projections with tanh (align_w folded into q)
  qk_kernel<<<MM / 16, 128, 0, stream>>>(px16, wq16, wk16, aw, qs16, k16);

  // 3. v = px @ Wv^T, written transposed per batch: vT16[b][c][n]
  {
    GemmP g = {};
    g.A = px16; g.lda = CC; g.Bt = wv16; g.ldb = CC; g.K = CC; g.outT = vT16;
    wmma_gemm_k<EPI_VT><<<dim3(CC / 128, MM / 128), 256, 0, stream>>>(g);
  }

  hipMemsetAsync(bnSum, 0, 2 * CC * sizeof(float), stream);

  // 4. per batch: attnexp = exp(qs @ k^T) + rowsum; then (attnexp @ v)/rowsum + px
  for (int b = 0; b < BB; ++b) {
    hipMemsetAsync(rowsum, 0, NN * sizeof(float), stream);
    {
      GemmP g = {};
      g.A  = qs16 + (size_t)b * NN * PP; g.lda = PP;
      g.Bt = k16  + (size_t)b * NN * PP; g.ldb = PP;
      g.K = PP; g.outH = attnexp; g.ldo = NN; g.rowsum = rowsum;
      wmma_gemm_k<EPI_EXP><<<dim3(NN / 128, NN / 128), 256, 0, stream>>>(g);
    }
    {
      GemmP g = {};
      g.A  = attnexp; g.lda = NN;
      g.Bt = vT16 + (size_t)b * CC * NN; g.ldb = NN;
      g.K = NN; g.outH = c16 + (size_t)b * NN * CC; g.ldo = CC;
      g.rs = rowsum; g.xres = x + (size_t)b * CC * NN;
      wmma_gemm_k<EPI_AV><<<dim3(CC / 128, NN / 128), 256, 0, stream>>>(g);
    }
  }

  // 5. fc: content @ Wfc^T + bfc -> d_out (transposed [B][C][N]) + BN partial sums
  {
    GemmP g = {};
    g.A = c16; g.lda = CC; g.Bt = wfc16; g.ldb = CC; g.K = CC;
    g.outF = out; g.bias = bfc; g.bnSum = bnSum; g.bnSqs = bnSqs;
    wmma_gemm_k<EPI_FC><<<dim3(CC / 128, MM / 128), 256, 0, stream>>>(g);
  }

  // 6. batchnorm (training-mode batch stats) + relu, in place on d_out
  bn_finalize<<<(CC + 255) / 256, 256, 0, stream>>>(bnSum, bnSqs, gamma, beta,
                                                    bnScale, bnShift, 1.0f / (float)MM);
  size_t total = (size_t)BB * CC * NN;
  bn_apply<<<(int)((total + 255) / 256), 256, 0, stream>>>(out, bnScale, bnShift, total);
}
